// MHCACLayer_10282151707247
// MI455X (gfx1250) — matrix operand order repaired and checked by reading, not compiled, not run on hardware
//
#include <hip/hip_runtime.h>
#include <hip/hip_bf16.h>

// ---------------- common WMMA types / helpers ----------------
typedef __attribute__((ext_vector_type(16))) __bf16        v16bf;
typedef __attribute__((ext_vector_type(8)))  float         v8f;
typedef __attribute__((ext_vector_type(4)))  unsigned int  uint4v;

union Frag {
    v16bf        v;
    unsigned int u[8];
    uint4v       q[2];
};

__device__ __forceinline__ unsigned short f2bf(float f) {
    union { float f; unsigned int u; } c; c.f = f;
    unsigned int r = c.u + 0x7fffu + ((c.u >> 16) & 1u);   // round-nearest-even
    return (unsigned short)(r >> 16);
}
__device__ __forceinline__ unsigned int pack2(float lo, float hi) {
    return (unsigned int)f2bf(lo) | ((unsigned int)f2bf(hi) << 16);
}

// A-matrix 16x32 frag: lane m=lane&15, lane-half kh; dwords {4kh..4kh+3} and {8+4kh..}
__device__ __forceinline__ void loadFragA(Frag& f, const unsigned int* b, int kh) {
    f.q[0] = *(const uint4v*)(b + 4 * kh);
    f.q[1] = *(const uint4v*)(b + 8 + 4 * kh);
}
// B-matrix 32x16 frag: lane n=lane&15; dwords {4kh..4kh+3} and {8+4kh..} of the column's K-window
__device__ __forceinline__ void loadFragB(Frag& f, const unsigned int* b, int kh) {
    f.q[0] = *(const uint4v*)(b + 4 * kh);
    f.q[1] = *(const uint4v*)(b + 8 + 4 * kh);
}

#define WMMA_BF16(A, B, C) \
    __builtin_amdgcn_wmma_f32_16x16x32_bf16(false, (A), false, (B), (short)0, (C), false, false)

// ---------------- fp32 -> bf16 bulk convert (b128 in / b128 out) ----------------
__global__ __launch_bounds__(256)
void cvt_bf16_kernel(const float* __restrict__ in, unsigned short* __restrict__ out, long n)
{
    long i = ((long)blockIdx.x * 256 + threadIdx.x) * 8;
    if (i >= n) return;
    float4 a = *(const float4*)(in + i);
    float4 b = *(const float4*)(in + i + 4);
    uint4v o;
    o[0] = pack2(a.x, a.y);
    o[1] = pack2(a.z, a.w);
    o[2] = pack2(b.x, b.y);
    o[3] = pack2(b.z, b.w);
    *(uint4v*)(out + i) = o;
}

// ---------------- weight transpose + convert: W[K][N] f32 -> Wt[N][K] bf16 ----------------
__global__ __launch_bounds__(256)
void transpose_cvt_kernel(const float* __restrict__ W, long zW,
                          unsigned short* __restrict__ Wt, long zWt,
                          int K, int N)
{
    __shared__ float tile[32][33];
    const int z = blockIdx.z;
    W  += (long)z * zW;
    Wt += (long)z * zWt;
    const int n0 = blockIdx.x * 32, k0 = blockIdx.y * 32;
    const int tx = threadIdx.x & 31, ty = threadIdx.x >> 5;
    #pragma unroll
    for (int j = 0; j < 4; ++j)
        tile[ty + 8 * j][tx] = W[(long)(k0 + ty + 8 * j) * N + n0 + tx];
    __syncthreads();
    #pragma unroll
    for (int j = 0; j < 4; ++j)
        Wt[(long)(n0 + ty + 8 * j) * K + k0 + tx] = f2bf(tile[tx][ty + 8 * j]);
}

// ---------------- register-resident bf16 WMMA GEMM ----------------
// C[M,N] = Abf[M,K] (bf16 row-major) x Wt[N,K] (bf16, pre-transposed) + bias.
// BM=128 (4 rowgroups x 32), BN=64 (2 colgroups x 32); wave = 2x2 16x16 subtiles.
// K must be a multiple of 64. No LDS, no barriers: every frag = two b128 loads.
__global__ __launch_bounds__(256)
void gemm_bf_wmma(const unsigned short* __restrict__ Abf, long aZ, int lda,
                  const unsigned short* __restrict__ Wt, long wZ,
                  const float* __restrict__ bias, long biasZ,
                  void* __restrict__ outv, long oZ, int ldo,
                  int M, int N, int K, int flags)
{
    const int z = blockIdx.z;
    Abf  += (long)z * aZ;
    Wt   += (long)z * wZ;
    bias += (long)z * biasZ;

    const int col0 = blockIdx.x * 64;     // x = col blocks (fast) -> A panel L2 reuse
    const int row0 = blockIdx.y * 128;
    const int lane = threadIdx.x & 31;
    const int w    = threadIdx.x >> 5;    // 0..7
    const int rg   = w & 3;               // row subtile group (32 rows)
    const int cg   = w >> 2;              // col subtile group (32 cols)
    const int kh   = lane >> 4;
    const int m    = lane & 15;

    // clamp OOB rows to a valid row (stores are guarded; garbage rows never stored)
    int r0 = row0 + rg * 32 + m;       if (r0 >= M) r0 = M - 1;
    int r1 = row0 + rg * 32 + 16 + m;  if (r1 >= M) r1 = M - 1;

    const unsigned int* a0 = (const unsigned int*)(Abf + (long)r0 * lda);
    const unsigned int* a1 = (const unsigned int*)(Abf + (long)r1 * lda);
    const unsigned int* b0 = (const unsigned int*)(Wt + (long)(col0 + cg * 32 + m) * K);
    const unsigned int* b1 = (const unsigned int*)(Wt + (long)(col0 + cg * 32 + 16 + m) * K);

    v8f c00 = {}, c01 = {}, c10 = {}, c11 = {};

    for (int k0 = 0; k0 < K; k0 += 64) {
        if (k0 + 256 < K) {   // stream-ahead hint -> global_prefetch_b8
            __builtin_prefetch((const char*)a0 + (long)(k0 + 256) * 2, 0, 1);
            __builtin_prefetch((const char*)b0 + (long)(k0 + 256) * 2, 0, 1);
        }
        #pragma unroll
        for (int s = 0; s < 2; ++s) {
            const int kd = (k0 + s * 32) >> 1;   // dword offset of this 32-K window
            Frag fa0, fa1, fb0, fb1;
            loadFragA(fa0, a0 + kd, kh);
            loadFragA(fa1, a1 + kd, kh);
            loadFragB(fb0, b0 + kd, kh);
            loadFragB(fb1, b1 + kd, kh);
            c00 = WMMA_BF16(fa0.v, fb0.v, c00);
            c01 = WMMA_BF16(fa0.v, fb1.v, c01);
            c10 = WMMA_BF16(fa1.v, fb0.v, c10);
            c11 = WMMA_BF16(fa1.v, fb1.v, c11);
        }
    }

    const bool relu = flags & 1;
    const bool obf  = flags & 2;
    const int  n    = lane & 15;
    #pragma unroll
    for (int i = 0; i < 2; ++i) {
        #pragma unroll
        for (int j = 0; j < 2; ++j) {
            v8f cc = (i == 0) ? ((j == 0) ? c00 : c01) : ((j == 0) ? c10 : c11);
            const int gcol = col0 + cg * 32 + j * 16 + n;
            const float bv = bias[gcol];
            #pragma unroll
            for (int r = 0; r < 8; ++r) {
                const int grow = row0 + rg * 32 + i * 16 + kh * 8 + r;
                if (grow < M) {
                    float vv = cc[r] + bv;
                    if (relu) vv = fmaxf(vv, 0.0f);
                    const long idx = (long)z * oZ + (long)grow * ldo + gcol;
                    if (obf) ((unsigned short*)outv)[idx] = f2bf(vv);
                    else     ((float*)outv)[idx] = vv;
                }
            }
        }
    }
}

// ---------------- fused attention per (b,h) ----------------
#define HH 12
#define TT 14
#define MM 2048
#define HD64 64

__global__ __launch_bounds__(128)
void attention_kernel(const unsigned short* __restrict__ Qbf,
                      const unsigned short* __restrict__ Kbf,
                      const unsigned short* __restrict__ Vbf,
                      float* __restrict__ attnw,
                      float* __restrict__ attn_out)
{
    __shared__ float sc[16 * MM];               // 128 KB scores / probs (CDNA5: 320KB/WGP)
    __shared__ unsigned short vst[HD64 * 32];   // V block, transposed [col][k]
    __shared__ float red[8];

    const int h    = blockIdx.x;
    const int b    = blockIdx.y;
    const int tid  = threadIdx.x;
    const int lane = tid & 31;
    const int w    = tid >> 5;                  // wave 0..3
    const int m    = lane & 15;
    const int kh   = lane >> 4;
    const int n    = lane & 15;

    // ---- Q fragments (16x64 as two 16x32 frags); rows >= T zeroed ----
    Frag qa0, qa1;
    {
        const unsigned int* qp =
            (const unsigned int*)(Qbf + ((long)b * TT + (m < TT ? m : 0)) * 768 + h * HD64);
        if (m < TT) { loadFragA(qa0, qp, kh); loadFragA(qa1, qp + 16, kh); }
        else {
            uint4v zz = {};
            qa0.q[0] = zz; qa0.q[1] = zz; qa1.q[0] = zz; qa1.q[1] = zz;
        }
    }

    // ---- scores: wave w handles key blocks w, w+4, ... ----
    for (int kb = w; kb < MM / 16; kb += 4) {
        const int m0 = kb * 16;
        const unsigned int* kp =
            (const unsigned int*)(Kbf + ((long)b * MM + m0 + n) * 768 + h * HD64);
        Frag fk0, fk1;
        loadFragB(fk0, kp, kh);
        loadFragB(fk1, kp + 16, kh);
        v8f acc = {};
        acc = WMMA_BF16(qa0.v, fk0.v, acc);
        acc = WMMA_BF16(qa1.v, fk1.v, acc);
        #pragma unroll
        for (int r = 0; r < 8; ++r)
            sc[(kh * 8 + r) * MM + m0 + n] = acc[r] * 0.125f;  // 1/sqrt(64)
    }
    __syncthreads();

    // ---- softmax rows t<14; write fp32 attn_w to d_out ----
    for (int t = 0; t < TT; ++t) {
        float* srow = sc + t * MM;
        float lmax = -1e30f;
        for (int i = tid; i < MM; i += 128) lmax = fmaxf(lmax, srow[i]);
        #pragma unroll
        for (int off = 16; off > 0; off >>= 1)
            lmax = fmaxf(lmax, __shfl_xor(lmax, off, 32));
        if (lane == 0) red[w] = lmax;
        __syncthreads();
        const float rmax = fmaxf(fmaxf(red[0], red[1]), fmaxf(red[2], red[3]));
        __syncthreads();

        float lsum = 0.0f;
        for (int i = tid; i < MM; i += 128) {
            float e = __expf(srow[i] - rmax);
            srow[i] = e;
            lsum += e;
        }
        #pragma unroll
        for (int off = 16; off > 0; off >>= 1)
            lsum += __shfl_xor(lsum, off, 32);
        if (lane == 0) red[w] = lsum;
        __syncthreads();
        const float inv = 1.0f / (red[0] + red[1] + red[2] + red[3]);
        __syncthreads();

        float* wout = attnw + (((long)b * TT + t) * HH + h) * MM;
        for (int i = tid; i < MM; i += 128) {
            float pv = srow[i] * inv;
            srow[i] = pv;
            wout[i] = pv;
        }
    }
    __syncthreads();

    // ---- in-place fp32 -> packed bf16 probs (chunked; reads precede writes) ----
    unsigned int* pb32 = (unsigned int*)sc;
    for (int c = 0; c < 16; ++c) {
        const int base = c * 2048 + tid * 16;
        float vals[16];
        #pragma unroll
        for (int j = 0; j < 16; ++j) vals[j] = sc[base + j];
        __syncthreads();
        #pragma unroll
        for (int j = 0; j < 8; ++j) pb32[(base >> 1) + j] = pack2(vals[2 * j], vals[2 * j + 1]);
        __syncthreads();
    }

    // ---- P(16x2048) x V(2048x64): wave w owns hd columns [16w,16w+16) ----
    v8f oacc = {};
    const unsigned int* pbase = (const unsigned int*)sc;   // bf16 probs, row stride 1024 dwords
    for (int ks = 0; ks < MM / 32; ++ks) {
        __syncthreads();
        for (int i = tid; i < 2048; i += 128) {            // stage 32x64 V block transposed
            int r = i >> 6, cc = i & 63;
            vst[cc * 32 + r] = Vbf[((long)b * MM + ks * 32 + r) * 768 + h * HD64 + cc];
        }
        __syncthreads();
        Frag pa, vb;
        loadFragA(pa, pbase + m * 1024 + ks * 16, kh);
        loadFragB(vb, (const unsigned int*)vst + (w * 16 + n) * 16, kh);
        oacc = WMMA_BF16(pa.v, vb.v, oacc);
    }
    #pragma unroll
    for (int r = 0; r < 8; ++r) {
        const int row = kh * 8 + r;
        if (row < TT)
            attn_out[((long)b * TT + row) * 768 + h * HD64 + w * 16 + n] = oacc[r];
    }
}

// ---------------- fused residual + LayerNorm (dim=768), optional bf16 side-output ----------------
__global__ __launch_bounds__(256)
void add_ln_kernel(const float* __restrict__ a, const float* __restrict__ bsrc,
                   const float* __restrict__ g, const float* __restrict__ be,
                   float* __restrict__ out, unsigned short* __restrict__ outbf)
{
    __shared__ float red[16];
    const long row = blockIdx.x;
    const int  tid = threadIdx.x;
    const float* pa = a + row * 768;
    const float* pb = bsrc + row * 768;

    float v[3], s = 0.0f, s2 = 0.0f;
    #pragma unroll
    for (int j = 0; j < 3; ++j) {
        float x = pa[tid + 256 * j] + pb[tid + 256 * j];
        v[j] = x; s += x; s2 += x * x;
    }
    #pragma unroll
    for (int off = 16; off > 0; off >>= 1) {
        s  += __shfl_xor(s,  off, 32);
        s2 += __shfl_xor(s2, off, 32);
    }
    const int w = tid >> 5, lane = tid & 31;
    if (lane == 0) { red[w] = s; red[8 + w] = s2; }
    __syncthreads();
    if (tid == 0) {
        float ts = 0.0f, ts2 = 0.0f;
        for (int i = 0; i < 8; ++i) { ts += red[i]; ts2 += red[8 + i]; }
        red[0] = ts; red[1] = ts2;
    }
    __syncthreads();
    const float mean = red[0] * (1.0f / 768.0f);
    const float var  = red[1] * (1.0f / 768.0f) - mean * mean;
    const float rstd = rsqrtf(var + 1e-5f);
    #pragma unroll
    for (int j = 0; j < 3; ++j) {
        const int c = tid + 256 * j;
        const float y = (v[j] - mean) * rstd * g[c] + be[c];
        out[row * 768 + c] = y;
        if (outbf) outbf[row * 768 + c] = f2bf(y);
    }
}

// ---------------- host launch ----------------
extern "C" void kernel_launch(void* const* d_in, const int* in_sizes, int n_in,
                              void* d_out, int out_size, void* d_ws, size_t ws_size,
                              hipStream_t stream)
{
    (void)in_sizes; (void)n_in; (void)out_size; (void)ws_size;
    const float* tok = (const float*)d_in[0];   // [32,14,768]
    const float* kv  = (const float*)d_in[1];   // [32,2048,1024]
    const float* WQ  = (const float*)d_in[2];   // [14,768,768]
    const float* bQ  = (const float*)d_in[3];   // [14,768]
    const float* WK  = (const float*)d_in[4];   // [1024,768]
    const float* bK  = (const float*)d_in[5];
    const float* WV  = (const float*)d_in[6];
    const float* bV  = (const float*)d_in[7];
    const float* W1  = (const float*)d_in[8];   // [768,3072]
    const float* b1  = (const float*)d_in[9];
    const float* W2  = (const float*)d_in[10];  // [3072,768]
    const float* b2  = (const float*)d_in[11];
    const float* g1  = (const float*)d_in[12];
    const float* be1 = (const float*)d_in[13];
    const float* g2  = (const float*)d_in[14];
    const float* be2 = (const float*)d_in[15];

    const int B = 32, T = 14, M = 2048, E = 768, KV = 1024, FF = 3072;

    float* out_x = (float*)d_out;                          // [B,T,768]
    float* attnw = (float*)d_out + (long)B * T * E;        // [B,T,12,2048]

    // workspace carve-up
    char* ws = (char*)d_ws;
    unsigned short* Akv  = (unsigned short*)ws;  ws += (long)B * M * KV * 2;   // kv bf16
    unsigned short* tokb = (unsigned short*)ws;  ws += (long)B * T * E * 2;
    unsigned short* Qbf  = (unsigned short*)ws;  ws += (long)B * T * E * 2;
    unsigned short* Kbf  = (unsigned short*)ws;  ws += (long)B * M * E * 2;
    unsigned short* Vbf  = (unsigned short*)ws;  ws += (long)B * M * E * 2;
    unsigned short* WKt  = (unsigned short*)ws;  ws += (long)KV * E * 2;       // [768][1024]
    unsigned short* WVt  = (unsigned short*)ws;  ws += (long)KV * E * 2;
    unsigned short* W1t  = (unsigned short*)ws;  ws += (long)E * FF * 2;       // [3072][768]
    unsigned short* W2t  = (unsigned short*)ws;  ws += (long)E * FF * 2;       // [768][3072]
    unsigned short* WQt  = (unsigned short*)ws;  ws += (long)T * E * E * 2;    // [t][768][768]
    unsigned short* xbf  = (unsigned short*)ws;  ws += (long)B * T * E * 2;
    unsigned short* ffhb = (unsigned short*)ws;  ws += (long)B * T * FF * 2;
    float* attn_out = (float*)ws;                ws += (long)B * T * E * 4;
    float* xbuf     = (float*)ws;                ws += (long)B * T * E * 4;
    float* ffy      = (float*)ws;                ws += (long)B * T * E * 4;

    // activations fp32 -> bf16
    cvt_bf16_kernel<<<(long)B * M * KV / 8 / 256, 256, 0, stream>>>(kv, Akv, (long)B * M * KV);
    cvt_bf16_kernel<<<(long)B * T * E / 8 / 256, 256, 0, stream>>>(tok, tokb, (long)B * T * E);

    // weights -> bf16 [N][K]
    transpose_cvt_kernel<<<dim3(E / 32, KV / 32, 1), 256, 0, stream>>>(WK, 0, WKt, 0, KV, E);
    transpose_cvt_kernel<<<dim3(E / 32, KV / 32, 1), 256, 0, stream>>>(WV, 0, WVt, 0, KV, E);
    transpose_cvt_kernel<<<dim3(FF / 32, E / 32, 1), 256, 0, stream>>>(W1, 0, W1t, 0, E, FF);
    transpose_cvt_kernel<<<dim3(E / 32, FF / 32, 1), 256, 0, stream>>>(W2, 0, W2t, 0, FF, E);
    transpose_cvt_kernel<<<dim3(E / 32, E / 32, T), 256, 0, stream>>>(WQ, (long)E * E, WQt, (long)E * E, E, E);

    // K / V projections: [65536,1024] x [1024->768] -> bf16
    gemm_bf_wmma<<<dim3(E / 64, (B * M) / 128, 1), 256, 0, stream>>>(
        Akv, 0, KV, WKt, 0, bK, 0, Kbf, 0, E, B * M, E, KV, /*bf16 out*/2);
    gemm_bf_wmma<<<dim3(E / 64, (B * M) / 128, 1), 256, 0, stream>>>(
        Akv, 0, KV, WVt, 0, bV, 0, Vbf, 0, E, B * M, E, KV, 2);

    // Q projection, batched over t (per-z weight/bias/output strides)
    gemm_bf_wmma<<<dim3(E / 64, 1, T), 256, 0, stream>>>(
        tokb, (long)E, T * E,
        WQt, (long)E * E,
        bQ, (long)E,
        Qbf, (long)E, T * E,
        B, E, E, 2);

    // attention per (h, b)
    attention_kernel<<<dim3(HH, B), 128, 0, stream>>>(Qbf, Kbf, Vbf, attnw, attn_out);

    // x = LN(tok + attn_out)  (fp32 + bf16 side copy for FFN)
    add_ln_kernel<<<B * T, 256, 0, stream>>>(tok, attn_out, g1, be1, xbuf, xbf);

    // FFN (ffh produced directly as bf16 with ReLU fused)
    gemm_bf_wmma<<<dim3(FF / 64, (B * T + 127) / 128, 1), 256, 0, stream>>>(
        xbf, 0, E, W1t, 0, b1, 0, ffhb, 0, FF, B * T, FF, E, /*relu|bf16*/3);
    gemm_bf_wmma<<<dim3(E / 64, (B * T + 127) / 128, 1), 256, 0, stream>>>(
        ffhb, 0, FF, W2t, 0, b2, 0, ffy, 0, E, B * T, E, FF, 0);

    // out = LN(x + ff)
    add_ln_kernel<<<B * T, 256, 0, stream>>>(xbuf, ffy, g2, be2, out_x, (unsigned short*)nullptr);
}
